// VideoAttention_10548439679107
// MI455X (gfx1250) — matrix operand order repaired and checked by reading, not compiled, not run on hardware
//
#include <hip/hip_runtime.h>
#include <hip/hip_bf16.h>
#include <math.h>

// ---------------------------------------------------------------------------
// Types
// ---------------------------------------------------------------------------
typedef __attribute__((ext_vector_type(16))) _Float16 v16h;
typedef __attribute__((ext_vector_type(8)))  _Float16 v8h;
typedef __attribute__((ext_vector_type(4)))  _Float16 v4h;
typedef __attribute__((ext_vector_type(2)))  _Float16 v2h;
typedef __attribute__((ext_vector_type(8)))  float    v8f;
typedef __attribute__((ext_vector_type(4)))  unsigned v4u;
typedef __attribute__((ext_vector_type(8)))  unsigned v8u;

#define BATCH    4
#define SEQ      1568
#define DMODEL   768
#define NHEADS   12
#define HDIM     64
#define BHCOUNT  (BATCH * NHEADS)                 // 48
// fold softmax scale AND log2(e) into Q so attention runs in the exp2 domain
#define QSCALE   (0.125f * 1.4426950408889634f)

// ---------------------------------------------------------------------------
// Helpers
// ---------------------------------------------------------------------------
__device__ __forceinline__ v16h cat8(v8h lo, v8h hi) {
  return __builtin_shufflevector(lo, hi, 0,1,2,3,4,5,6,7,8,9,10,11,12,13,14,15);
}

// Single-instruction max: opaque to LLVM's fcanonicalize, so no v_max x,x
// operand-canonicalization pairs get inserted around it.
__device__ __forceinline__ float amax(float a, float b) {
  float d;
  asm("v_max_num_f32_e32 %0, %1, %2" : "=v"(d) : "v"(a), "v"(b));
  return d;
}

// Fused DPP16 + op reductions over each 16-lane half-wave: one VALU op per
// step.
__device__ __forceinline__ float row16_max(float v) {
  asm("v_max_num_f32_dpp %0, %1, %1 quad_perm:[1,0,3,2] row_mask:0xf bank_mask:0xf bound_ctrl:1"
      : "=v"(v) : "v"(v));
  asm("v_max_num_f32_dpp %0, %1, %1 quad_perm:[2,3,0,1] row_mask:0xf bank_mask:0xf bound_ctrl:1"
      : "=v"(v) : "v"(v));
  asm("v_max_num_f32_dpp %0, %1, %1 row_half_mirror row_mask:0xf bank_mask:0xf bound_ctrl:1"
      : "=v"(v) : "v"(v));
  asm("v_max_num_f32_dpp %0, %1, %1 row_mirror row_mask:0xf bank_mask:0xf bound_ctrl:1"
      : "=v"(v) : "v"(v));
  return v;
}
__device__ __forceinline__ float row16_sum(float v) {
  asm("v_add_f32_dpp %0, %1, %1 quad_perm:[1,0,3,2] row_mask:0xf bank_mask:0xf bound_ctrl:1"
      : "=v"(v) : "v"(v));
  asm("v_add_f32_dpp %0, %1, %1 quad_perm:[2,3,0,1] row_mask:0xf bank_mask:0xf bound_ctrl:1"
      : "=v"(v) : "v"(v));
  asm("v_add_f32_dpp %0, %1, %1 row_half_mirror row_mask:0xf bank_mask:0xf bound_ctrl:1"
      : "=v"(v) : "v"(v));
  asm("v_add_f32_dpp %0, %1, %1 row_mirror row_mask:0xf bank_mask:0xf bound_ctrl:1"
      : "=v"(v) : "v"(v));
  return v;
}

__device__ __forceinline__ v8f wmma_f16(v16h a, v16h b, v8f c) {
  return __builtin_amdgcn_wmma_f32_16x16x32_f16(false, a, false, b, (short)0, c,
                                                false, false);
}

// ---------------------------------------------------------------------------
// Tensor Data Mover: 2D f16 tile (tile_d0 x tile_d1) global->LDS with row pad.
//   pic: pad after 2^(pic+1) dwords; pac: pac+1 dwords of padding.
// Issued per-wave; completions retire in order -> double-buffer with
// s_wait_tensorcnt(1).
// ---------------------------------------------------------------------------
__device__ __forceinline__ void tdm_load_2d_f16(const void* lds_dst,
                                                const void* gptr,
                                                unsigned tile_d0,
                                                unsigned tile_d1,
                                                unsigned stride_elems,
                                                unsigned pic, unsigned pac) {
  const unsigned long long ga = (unsigned long long)(uintptr_t)gptr;
  const unsigned lds_off = (unsigned)(uintptr_t)lds_dst;
  const v4u g0 = {1u,                                    // count=1 (valid D#)
                  lds_off,                               // lds_addr
                  (unsigned)ga,                          // global_addr[31:0]
                  (unsigned)(ga >> 32) | (2u << 30)};    // addr[56:32] | type=2
  const unsigned dw0 = (1u << 16)      /* data_size = 2 bytes */
                     | (1u << 20)      /* pad_enable */
                     | (pic << 22) | (pac << 25);
  const v8u g1 = {dw0,
                  tile_d0 << 16,                  // tensor_dim0[15:0]  (=tile)
                  tile_d1 << 16,                  // tensor_dim1[15:0]  (=tile)
                  tile_d0 << 16,                  // tile_dim0
                  tile_d1,                        // tile_dim1 (tile_dim2=0)
                  stride_elems,                   // tensor_dim0_stride[31:0]
                  0u, 0u};
  asm volatile("tensor_load_to_lds %0, %1" :: "s"(g0), "s"(g1) : "memory");
}

// ---------------------------------------------------------------------------
// Kernel 0: f32 -> f16 convert (4-wide)
// ---------------------------------------------------------------------------
__global__ void cvt_f32_to_f16(const float* __restrict__ src,
                               _Float16* __restrict__ dst, int n4) {
  int i = blockIdx.x * blockDim.x + threadIdx.x;
  if (i < n4) {
    float4 v = ((const float4*)src)[i];
    v4h h = {(_Float16)v.x, (_Float16)v.y, (_Float16)v.z, (_Float16)v.w};
    ((v4h*)dst)[i] = h;
  }
}

// ---------------------------------------------------------------------------
// Tiled WMMA GEMM:  C[M,N] = A[M,K](f16) * B[K,N](f16) + bias
//   BM=128 BN=128 BK=32, 256 threads (8 waves, 4x2 wave grid, 32x64 per wave)
//   Double-buffered: TDM loads A tile t+1 while WMMAs run on tile t.
// MODE 0: QKV epilogue -> scatter f16 into Q/K/V [B,H,N,64]; Q pre-scaled.
// MODE 1: plain epilogue -> fp32 Out[row*N + col] = acc + bias[col]
// ---------------------------------------------------------------------------
#define BM   128
#define BN   128
#define BK   32
#define LDAS 40   // A tile LDS stride (halves): 64B data + 16B pad
#define LDBS 40

template <int MODE>
__global__ __launch_bounds__(256)
void gemm_wmma(const _Float16* __restrict__ A, const _Float16* __restrict__ B,
               const float* __restrict__ bias, void* __restrict__ Out,
               int M, int N, int K, int lda, int ldb) {
  __shared__ _Float16 As[2][BM * LDAS];   // [m][k], TDM-padded rows
  __shared__ _Float16 Bs[2][BN * LDBS];   // [n][k] (transposed)

  const int tid  = threadIdx.x;
  const int lane = tid & 31;
  const int wv   = tid >> 5;
  const int wm   = wv >> 1;               // 0..3  (32 rows each)
  const int wn   = wv & 1;                // 0..1  (64 cols each)
  const int lm   = lane & 15;
  const int lh   = lane >> 4;

  const int rowBase = blockIdx.x * BM;
  const int colBase = blockIdx.y * BN;

  v8f acc[2][4];
#pragma unroll
  for (int i = 0; i < 2; ++i)
#pragma unroll
    for (int j = 0; j < 4; ++j) acc[i][j] = (v8f)0.0f;

  // B staging plan: thread owns k-row pair {2*(tid>>4), +1}, 8 n-columns
  const int bkp = (tid >> 4) * 2;         // 0,2,..,30
  const int bns = (tid & 15) * 8;         // 0..120

  auto issueA = [&](int kt, int buf) {
    if (wv == 0)
      tdm_load_2d_f16(&As[buf][0], A + (size_t)rowBase * lda + kt,
                      /*tile_d0=*/BK, /*tile_d1=*/BM, (unsigned)lda,
                      /*pic=*/3, /*pac=*/3);  // pad 4 dw every 16 dw -> 80B
  };
  auto stageB = [&](int kt, int buf) {
    const _Float16* bg = B + (size_t)(kt + bkp) * ldb + colBase + bns;
    v8h b0 = *(const v8h*)bg;
    v8h b1 = *(const v8h*)(bg + ldb);
#pragma unroll
    for (int i = 0; i < 8; ++i) {
      v2h t = {b0[i], b1[i]};
      *(v2h*)&Bs[buf][(bns + i) * LDBS + bkp] = t;
    }
    if (kt + BK < K)
      __builtin_prefetch(bg + (size_t)BK * ldb, 0, 1);  // global_prefetch_b8
  };

  const int T = K / BK;
  issueA(0, 0);
  stageB(0, 0);

  for (int t = 0; t < T; ++t) {
    const int cur = t & 1, nxt = cur ^ 1;
    __syncthreads();                  // everyone done reading tile t-1 bufs
    if (t + 1 < T) {
      issueA((t + 1) * BK, nxt);      // DMA overlaps this step's compute
      stageB((t + 1) * BK, nxt);
    }
    if (wv == 0) {
      if (t + 1 < T) __builtin_amdgcn_s_wait_tensorcnt(1);  // tile t retired
      else           __builtin_amdgcn_s_wait_tensorcnt(0);
    }
    __syncthreads();                  // As[cur]/Bs[cur] visible to all

    // fragments (ISA 7.12.2 layouts)
    v16h af[2], bf[4];
#pragma unroll
    for (int i = 0; i < 2; ++i) {
      const int r = wm * 32 + i * 16 + lm;
      const _Float16* p = &As[cur][r * LDAS + lh * 8];
      af[i] = cat8(*(const v8h*)p, *(const v8h*)(p + 16));
    }
#pragma unroll
    for (int i = 0; i < 4; ++i) {
      const int n = wn * 64 + i * 16 + lm;
      const _Float16* p = &Bs[cur][n * LDBS + lh * 8];
      bf[i] = cat8(*(const v8h*)p, *(const v8h*)(p + 16));
    }
#pragma unroll
    for (int mi = 0; mi < 2; ++mi)
#pragma unroll
      for (int ni = 0; ni < 4; ++ni)
        acc[mi][ni] = wmma_f16(af[mi], bf[ni], acc[mi][ni]);
  }

  // --- epilogue ---
#pragma unroll
  for (int mi = 0; mi < 2; ++mi) {
#pragma unroll
    for (int ni = 0; ni < 4; ++ni) {
      const int col = colBase + wn * 64 + ni * 16 + lm;
      const float bv = bias[col];
      if (MODE == 0) {
        const int three = col / DMODEL;
        const int rem   = col % DMODEL;
        const int hh    = rem >> 6;
        const int dd    = rem & 63;
        const float sc  = (three == 0) ? QSCALE : 1.0f;  // pre-scale Q
        _Float16* base =
            (_Float16*)Out + (size_t)three * ((size_t)BHCOUNT * SEQ * HDIM);
#pragma unroll
        for (int j = 0; j < 8; ++j) {
          const int row = rowBase + wm * 32 + mi * 16 + j + 8 * lh;
          const int b = row / SEQ, n = row % SEQ;
          base[((size_t)(b * NHEADS + hh) * SEQ + n) * HDIM + dd] =
              (_Float16)((acc[mi][ni][j] + bv) * sc);
        }
      } else {
        float* outp = (float*)Out;
#pragma unroll
        for (int j = 0; j < 8; ++j) {
          const int row = rowBase + wm * 32 + mi * 16 + j + 8 * lh;
          outp[(size_t)row * N + col] = acc[mi][ni][j] + bv;
        }
      }
    }
  }
}

// ---------------------------------------------------------------------------
// Flash attention core. Grid: (13, 48), 256 threads (8 waves).
// Wave w owns Q rows [blk*128 + w*16, +16). 49 KV steps of 32 keys.
// Double-buffered: TDM loads K tile kb+1 (and V stage) during step kb.
// Scores arrive pre-scaled by 1/8*log2(e); softmax in exp2 domain.
// ---------------------------------------------------------------------------
#define KSLD 72   // Ks stride (halves): 128B data + 16B pad
#define VSLD 40   // Vs / P stride: 80B rows
#define KBLK (SEQ / 32)                  // 49

__global__ __launch_bounds__(256)
void attn_flash(const _Float16* __restrict__ Qf, const _Float16* __restrict__ Kf,
                const _Float16* __restrict__ Vf, _Float16* __restrict__ O16) {
  __shared__ _Float16 Ks[2][32 * KSLD];    // [key][k], TDM-padded
  __shared__ _Float16 Vs[2][64 * VSLD];    // [d][key] transposed
  __shared__ _Float16 Pl[8][16 * VSLD];    // per-wave P staging [row][key]

  const int tid  = threadIdx.x;
  const int lane = tid & 31;
  const int wv   = tid >> 5;
  const int lm   = lane & 15;
  const int lh   = lane >> 4;

  const int bh = blockIdx.y;
  const int b  = bh / NHEADS;
  const int hh = bh % NHEADS;
  const size_t base = (size_t)bh * SEQ * HDIM;
  const int rowBase = blockIdx.x * 128 + wv * 16;

  // Q as two A-fragments (rows clamped; OOB waves duplicate work, skip stores)
  v16h qf[2];
  {
    const int r = rowBase + lm;
    const int rc = (r < SEQ) ? r : (SEQ - 1);
    const _Float16* qp = Qf + base + (size_t)rc * HDIM;
#pragma unroll
    for (int c = 0; c < 2; ++c) {
      const _Float16* p = qp + c * 32 + lh * 8;
      qf[c] = cat8(*(const v8h*)p, *(const v8h*)(p + 16));
    }
  }

  v8f acc[4];
#pragma unroll
  for (int i = 0; i < 4; ++i) acc[i] = (v8f)0.0f;
  float mst[8], lst[8];
#pragma unroll
  for (int j = 0; j < 8; ++j) { mst[j] = -INFINITY; lst[j] = 0.0f; }

  // V staging plan: thread owns key pair {2*(tid>>4), +1}, 4 head-dims
  const int vkp = (tid >> 4) * 2;          // 0..30
  const int vds = (tid & 15) * 4;          // 0..60

  auto issueK = [&](int kb, int buf) {
    if (wv == 0)
      tdm_load_2d_f16(&Ks[buf][0], Kf + base + (size_t)(kb * 32) * HDIM,
                      /*tile_d0=*/HDIM, /*tile_d1=*/32, /*stride=*/HDIM,
                      /*pic=*/4, /*pac=*/3);  // pad 4 dw every 32 dw -> 144B
  };
  auto stageV = [&](int kb, int buf) {
    const _Float16* vp = Vf + base + (size_t)(kb * 32 + vkp) * HDIM + vds;
    v4h a0 = *(const v4h*)vp;
    v4h a1 = *(const v4h*)(vp + HDIM);
#pragma unroll
    for (int i = 0; i < 4; ++i) {
      v2h t = {a0[i], a1[i]};
      *(v2h*)&Vs[buf][(vds + i) * VSLD + vkp] = t;
    }
  };

  issueK(0, 0);
  stageV(0, 0);

  for (int kb = 0; kb < KBLK; ++kb) {
    const int cur = kb & 1, nxt = cur ^ 1;
    __syncthreads();
    if (kb + 1 < KBLK) {
      issueK(kb + 1, nxt);               // DMA overlaps this step's compute
      stageV(kb + 1, nxt);
    }
    if (wv == 0) {
      if (kb + 1 < KBLK) __builtin_amdgcn_s_wait_tensorcnt(1);
      else               __builtin_amdgcn_s_wait_tensorcnt(0);
    }
    __syncthreads();

    // --- S = Q K^T (two 16x16 tiles; K-dim 64 split into 2 fragments) ---
    v8f S[2];
#pragma unroll
    for (int t = 0; t < 2; ++t) {
      S[t] = (v8f)0.0f;
#pragma unroll
      for (int c = 0; c < 2; ++c) {
        const _Float16* p = &Ks[cur][(t * 16 + lm) * KSLD + c * 32 + lh * 8];
        v16h kbf = cat8(*(const v8h*)p, *(const v8h*)(p + 16));
        S[t] = wmma_f16(qf[c], kbf, S[t]);
      }
    }

    // --- online softmax (exp2 domain, fused DPP reductions) ---
    float alpha[8];
#pragma unroll
    for (int j = 0; j < 8; ++j) {
      const float s0 = S[0][j];
      const float s1 = S[1][j];
      const float mc = row16_max(amax(s0, s1));
      const float mn = amax(mst[j], mc);
      alpha[j] = exp2f(mst[j] - mn);
      const float p0 = exp2f(s0 - mn);
      const float p1 = exp2f(s1 - mn);
      lst[j] = lst[j] * alpha[j] + row16_sum(p0 + p1);
      mst[j] = mn;
      S[0][j] = p0;
      S[1][j] = p1;
    }
#pragma unroll
    for (int dt = 0; dt < 4; ++dt)
#pragma unroll
      for (int j = 0; j < 8; ++j) acc[dt][j] *= alpha[j];

    // --- transpose P: C-layout -> A-layout via per-wave LDS ---
#pragma unroll
    for (int j = 0; j < 8; ++j) {
      const int r = j + 8 * lh;
      Pl[wv][r * VSLD + lm]      = (_Float16)S[0][j];
      Pl[wv][r * VSLD + 16 + lm] = (_Float16)S[1][j];
    }
    asm volatile("" ::: "memory");  // wave-synchronous LDS: keep order
    v16h pa;
    {
      const _Float16* p = &Pl[wv][lm * VSLD + lh * 8];
      pa = cat8(*(const v8h*)p, *(const v8h*)(p + 16));
    }

    // --- O += P V ---
#pragma unroll
    for (int dt = 0; dt < 4; ++dt) {
      const _Float16* p = &Vs[cur][(dt * 16 + lm) * VSLD + lh * 8];
      v16h vb = cat8(*(const v8h*)p, *(const v8h*)(p + 16));
      acc[dt] = wmma_f16(pa, vb, acc[dt]);
    }
  }

  // --- normalize + store O in [B,N,D] f16 for the projection GEMM ---
  float invl[8];
#pragma unroll
  for (int j = 0; j < 8; ++j) invl[j] = 1.0f / lst[j];
#pragma unroll
  for (int dt = 0; dt < 4; ++dt)
#pragma unroll
    for (int j = 0; j < 8; ++j) {
      const int n = rowBase + j + 8 * lh;
      if (n < SEQ)
        O16[(size_t)(b * SEQ + n) * DMODEL + hh * HDIM + dt * 16 + lm] =
            (_Float16)(acc[dt][j] * invl[j]);
    }
}

// ---------------------------------------------------------------------------
// Host: orchestrate the pipeline (all on `stream`, graph-capture safe)
// ---------------------------------------------------------------------------
extern "C" void kernel_launch(void* const* d_in, const int* in_sizes, int n_in,
                              void* d_out, int out_size, void* d_ws,
                              size_t ws_size, hipStream_t stream) {
  const float* x      = (const float*)d_in[0];  // [4,1568,768]
  const float* qkv_w  = (const float*)d_in[1];  // [768,2304]
  const float* qkv_b  = (const float*)d_in[2];  // [2304]
  const float* proj_w = (const float*)d_in[3];  // [768,768]
  const float* proj_b = (const float*)d_in[4];  // [768]
  float* out = (float*)d_out;                   // [4,1568,768]

  const size_t XN = (size_t)BATCH * SEQ * DMODEL;
  const size_t WN = (size_t)DMODEL * 3 * DMODEL;
  const size_t PN = (size_t)DMODEL * DMODEL;
  const size_t QN = (size_t)BHCOUNT * SEQ * HDIM;

  char* w = (char*)d_ws;
  _Float16* Xh = (_Float16*)w;            w += XN * 2;
  _Float16* Wh = (_Float16*)w;            w += WN * 2;
  _Float16* Ph = (_Float16*)w;            w += PN * 2;
  _Float16* Qh = (_Float16*)w;            w += QN * 2;   // Q/K/V contiguous
  _Float16* Kh = (_Float16*)w;            w += QN * 2;
  _Float16* Vh = (_Float16*)w;            w += QN * 2;
  _Float16* Oh = (_Float16*)w;            w += XN * 2;
  (void)ws_size; (void)n_in; (void)in_sizes; (void)out_size;

  // 1) precision convert
  {
    const int T = 256;
    int n4 = (int)(XN / 4);
    cvt_f32_to_f16<<<(n4 + T - 1) / T, T, 0, stream>>>(x, Xh, n4);
    n4 = (int)(WN / 4);
    cvt_f32_to_f16<<<(n4 + T - 1) / T, T, 0, stream>>>(qkv_w, Wh, n4);
    n4 = (int)(PN / 4);
    cvt_f32_to_f16<<<(n4 + T - 1) / T, T, 0, stream>>>(proj_w, Ph, n4);
  }

  // 2) fused QKV projection + head transpose (Q pre-scaled by 1/8*log2e)
  gemm_wmma<0><<<dim3((BATCH * SEQ) / BM, (3 * DMODEL) / BN), 256, 0, stream>>>(
      Xh, Wh, qkv_b, (void*)Qh, BATCH * SEQ, 3 * DMODEL, DMODEL, DMODEL,
      3 * DMODEL);

  // 3) flash attention, output f16 [B,N,D]
  attn_flash<<<dim3((SEQ + 127) / 128, BHCOUNT), 256, 0, stream>>>(Qh, Kh, Vh,
                                                                   Oh);

  // 4) output projection, fp32 result
  gemm_wmma<1><<<dim3((BATCH * SEQ) / BM, DMODEL / BN), 256, 0, stream>>>(
      Oh, Ph, proj_b, (void*)out, BATCH * SEQ, DMODEL, DMODEL, DMODEL, DMODEL);
}
